// NodeAlignNodeLossConsistency_18416819765601
// MI455X (gfx1250) — matrix operand order repaired and checked by reading, not compiled, not run on hardware
//
#include <hip/hip_runtime.h>
#include <cstdint>
#include <cstddef>

// ---------------------------------------------------------------------------
// Types / helpers
// ---------------------------------------------------------------------------
typedef __bf16 bf16_t;
typedef __attribute__((ext_vector_type(16))) __bf16 v16bf;
typedef __attribute__((ext_vector_type(8)))  float  v8f;

#define DEV static __device__ __forceinline__

DEV unsigned short f2bf_bits(float f) {
  union { float f; unsigned int u; } x; x.f = f;
  unsigned int u = x.u;
  return (unsigned short)((u + 0x7FFFu + ((u >> 16) & 1u)) >> 16);   // RNE
}
DEV bf16_t f2bf(float f) {
  union { unsigned short s; bf16_t h; } y; y.s = f2bf_bits(f); return y.h;
}

union V16U { v16bf v; unsigned short s[16]; unsigned int u[8]; };

DEV float block_reduce_sum(float v, float* sdata) {
  int tid = threadIdx.x;
  for (int o = 16; o; o >>= 1) v += __shfl_xor(v, o, 32);
  if ((tid & 31) == 0) sdata[tid >> 5] = v;
  __syncthreads();
  int nw = blockDim.x >> 5;
  float r = (tid < nw) ? sdata[tid] : 0.f;
  if (tid < 32) for (int o = 16; o; o >>= 1) r += __shfl_xor(r, o, 32);
  return r;  // valid in thread 0
}

// Problem constants (from the reference)
static constexpr int   B_   = 256;
static constexpr int   NPG_ = 28;
static constexpr int   EPG_ = 64;
static constexpr int   D_   = 128;
static constexpr long  G_   = 512;          // 2*B
static constexpr long  NN_  = G_ * NPG_;    // 14336
static constexpr long  EE_  = G_ * EPG_;    // 32768
static constexpr float TEMP_ = 0.1f;
static constexpr float CW_   = 0.9f;

// ---------------------------------------------------------------------------
// Weight prep: f32 row-major [K,N] -> bf16 in WMMA B-operand swizzle.
// B layout (16-bit, 32x16 per K-tile): lane L holds N = L%16, K half = L/16,
// element i <-> K = 16*(L/16) + i.  Block (kt,nt) is a contiguous 512-elem
// chunk; lane L reads 16 consecutive bf16 at block + L*16.
// ---------------------------------------------------------------------------
__global__ void prep_w(const float* __restrict__ src, bf16_t* __restrict__ dst,
                       int K, int Ncols) {
  int i = blockIdx.x * 256 + threadIdx.x;
  if (i >= K * Ncols) return;
  int k = i / Ncols, n = i % Ncols;
  int kt = k >> 5, kk = k & 31, nt = n >> 4, ln = n & 15;
  int lane = ln + 16 * (kk >> 4);
  int ii = kk & 15;
  size_t d = ((size_t)(kt * (Ncols >> 4) + nt) * 32 + lane) * 16 + ii;
  dst[d] = f2bf(src[i]);
}

// ---------------------------------------------------------------------------
// Generic bf16 WMMA GEMM: D[M,Ncols] = epi(A[M,K] @ B[K,Ncols] + bias)
//   ASRC 0: A = f32 contiguous rows (converted on load)
//        1: A = bf16 contiguous rows
//        2: A = concat( hb[idxA[r]], hb[idxB[r]], eb[r] )   (K = 384)
//        3: A = concat( aB0[r], aB1[r] )                    (K = 256)
//   EPI  0: outF = v
//        1: outBf = v
//        2: outF += v
//        3: outF += v ; outBf = bf16(outF)   (residual h update)
//        4: outF = v ; outBf = bf16(v)
//   One wave per 16-row tile; 4 waves per block. All modes compile-time.
// ---------------------------------------------------------------------------
template <int ASRC, int EPI, int RELU, int KT>
__global__ __launch_bounds__(128) void gemm_k(
    const float*  __restrict__ aF,
    const bf16_t* __restrict__ aB0,
    const bf16_t* __restrict__ aB1,
    const int*    __restrict__ idxA,
    const int*    __restrict__ idxB,
    const bf16_t* __restrict__ W,
    const float*  __restrict__ bias,
    float*        __restrict__ outF,
    bf16_t*       __restrict__ outBf,
    int Ncols) {
  constexpr int K = KT * 32;
  const int tid  = threadIdx.x;
  const int lane = tid & 31;
  const int wave = tid >> 5;
  const long rows = ((long)blockIdx.x * 4 + wave) * 16;
  const int  l  = lane & 15;
  const int  hi = lane >> 4;
  const long r  = rows + l;   // A row owned by this lane

  long iA = 0, iB = 0;
  if (ASRC == 2) { iA = idxA[r]; iB = idxB[r]; }

  // ---- load A tiles (16x32 bf16; element s <-> K = 16*hi + s) ----
  v16bf A[KT];
#pragma unroll
  for (int kt = 0; kt < KT; ++kt) {
    V16U u;
#pragma unroll
    for (int p = 0; p < 8; ++p) {
      int kloc = 2 * p + 16 * hi;   // K within tile
      int kg = kt * 32 + kloc;                          // global K
      if (ASRC == 0) {
        const float* s = aF + r * K + kg;
        u.s[2 * p]     = f2bf_bits(s[0]);
        u.s[2 * p + 1] = f2bf_bits(s[1]);
      } else if (ASRC == 1) {
        u.u[p] = *(const unsigned int*)(aB0 + r * K + kg);
      } else if (ASRC == 2) {
        const bf16_t* s;
        if (kg < 128)       s = aB0 + iA * 128 + kg;
        else if (kg < 256)  s = aB0 + iB * 128 + (kg - 128);
        else                s = aB1 + r * 128 + (kg - 256);
        u.u[p] = *(const unsigned int*)s;
      } else {  // ASRC == 3
        const bf16_t* s = (kg < 128) ? (aB0 + r * 128 + kg)
                                     : (aB1 + r * 128 + (kg - 128));
        u.u[p] = *(const unsigned int*)s;
      }
    }
    A[kt] = u.v;
  }

  // Per-v output row base (row = rows + v + 8*hi), hoisted out of nt loop.
  long rbase[8];
#pragma unroll
  for (int v = 0; v < 8; ++v) rbase[v] = (rows + v + 8 * hi) * (long)Ncols;

  const int NT = Ncols >> 4;
  for (int nt = 0; nt < NT; ++nt) {
    if (nt + 1 < NT)   // prefetch next N-tile's first B block
      __builtin_prefetch(W + ((long)(nt + 1) * 32 + lane) * 16, 0, 1);
    v8f c = {};
#pragma unroll
    for (int kt = 0; kt < KT; ++kt) {
      const v16bf b = *(const v16bf*)(W + (((long)(kt * NT + nt)) * 32 + lane) * 16);
      c = __builtin_amdgcn_wmma_f32_16x16x32_bf16(
              /*neg_a=*/false, A[kt], /*neg_b=*/false, b,
              /*c_mod=*/(short)0, c, /*reuse_a=*/false, /*reuse_b=*/false);
    }
    const int n = nt * 16 + l;
    const float bn = bias[n];
#pragma unroll
    for (int v = 0; v < 8; ++v) {
      float val = c[v] + bn;
      if (RELU) val = fmaxf(val, 0.f);
      const long oi = rbase[v] + n;
      if (EPI == 0) {
        outF[oi] = val;
      } else if (EPI == 1) {
        outBf[oi] = f2bf(val);
      } else if (EPI == 2) {
        outF[oi] += val;
      } else if (EPI == 3) {
        float nh = outF[oi] + val; outF[oi] = nh; outBf[oi] = f2bf(nh);
      } else {  // EPI == 4
        outF[oi] = val; outBf[oi] = f2bf(val);
      }
    }
  }
}

// ---------------------------------------------------------------------------
// Deterministic per-graph segment-sum:  agg[n] = sum_{to==n} f + sum_{from==n} r
// One block per graph, thread = feature column, LDS accumulators (no atomics).
// ---------------------------------------------------------------------------
__global__ __launch_bounds__(128) void aggregate_k(
    const float* __restrict__ f, const float* __restrict__ r,
    const int* __restrict__ from, const int* __restrict__ to,
    bf16_t* __restrict__ aggb) {
  const int g = blockIdx.x, d = threadIdx.x;
  __shared__ float acc[NPG_ * D_];
#pragma unroll
  for (int n = 0; n < NPG_; ++n) acc[n * D_ + d] = 0.f;
  for (int e = 0; e < EPG_; ++e) {
    const long ge = (long)g * EPG_ + e;
    const int tl = to[ge]   - g * NPG_;
    const int fl = from[ge] - g * NPG_;
    acc[tl * D_ + d] += f[ge * D_ + d];
    acc[fl * D_ + d] += r[ge * D_ + d];
  }
  for (int n = 0; n < NPG_; ++n)
    aggb[((long)g * NPG_ + n) * D_ + d] = f2bf(acc[n * D_ + d]);
}

// Zero-padded [G,32,128] bf16 node tensor for the sinkhorn MLP.
__global__ __launch_bounds__(128) void pad_k(const bf16_t* __restrict__ hb,
                                             bf16_t* __restrict__ hs) {
  const int row = blockIdx.x, d = threadIdx.x;
  const int g = row >> 5, p = row & 31;
  hs[(long)row * D_ + d] = (p < NPG_) ? hb[((long)g * NPG_ + p) * D_ + d] : f2bf(0.f);
}

// scores[b,q,c] = <mq[q], mc[c]> with size masking (t: [G,32,64] f32)
__global__ __launch_bounds__(256) void scores_node_k(
    const float* __restrict__ t, const int* __restrict__ qs,
    const int* __restrict__ cs, float* __restrict__ sn) {
  const int b = blockIdx.x, tid = threadIdx.x;
  __shared__ float sq[32 * 64];
  __shared__ float sc[32 * 65];     // padded rows: conflict-free
  const int qn = qs[b], cn = cs[b];
  for (int x = tid; x < 2048; x += 256) {
    int p = x >> 6, d = x & 63;
    sq[p * 64 + d] = (p < qn) ? t[((long)(2 * b) * 32 + p) * 64 + d] : 0.f;
    sc[p * 65 + d] = (p < cn) ? t[((long)(2 * b + 1) * 32 + p) * 64 + d] : 0.f;
  }
  __syncthreads();
  for (int x = tid; x < 1024; x += 256) {
    int q = x >> 5, c = x & 31;
    float a = 0.f;
#pragma unroll 8
    for (int d = 0; d < 64; ++d) a += sq[q * 64 + d] * sc[c * 65 + d];
    sn[(long)b * 1024 + x] = a;
  }
}

// Sinkhorn on 32x32: one wave per batch, lane owns column c, rows in VGPRs.
__global__ __launch_bounds__(32) void sinkhorn_node_k(
    const float* __restrict__ sn, float* __restrict__ plan) {
  const int b = blockIdx.x, lane = threadIdx.x;
  float la[32];
#pragma unroll
  for (int q = 0; q < 32; ++q)
    la[q] = sn[(long)b * 1024 + q * 32 + lane] * (1.f / TEMP_);
  for (int it = 0; it < 20; ++it) {
#pragma unroll
    for (int q = 0; q < 32; ++q) {            // axis=2 (over c): cross-lane lse
      float v = la[q], m = v;
      for (int o = 16; o; o >>= 1) m = fmaxf(m, __shfl_xor(m, o, 32));
      float s = expf(v - m);
      for (int o = 16; o; o >>= 1) s += __shfl_xor(s, o, 32);
      la[q] = v - (m + logf(s));
    }
    float m2 = -1e30f;                        // axis=1 (over q): in-lane lse
#pragma unroll
    for (int q = 0; q < 32; ++q) m2 = fmaxf(m2, la[q]);
    float s2 = 0.f;
#pragma unroll
    for (int q = 0; q < 32; ++q) s2 += expf(la[q] - m2);
    const float l2 = m2 + logf(s2);
#pragma unroll
    for (int q = 0; q < 32; ++q) la[q] -= l2;
  }
#pragma unroll
  for (int q = 0; q < 32; ++q)
    plan[(long)b * 1024 + q * 32 + lane] = expf(la[q]);
}

// out[b] = -sum relu(q_nodes - plan @ c_nodes)
__global__ __launch_bounds__(128) void node_align_k(
    const float* __restrict__ plan, const float* __restrict__ h,
    float* __restrict__ out) {
  const int b = blockIdx.x, tid = threadIdx.x;
  __shared__ float pl[1024];
  __shared__ float cn[32 * D_];
  __shared__ float red[4];
  for (int x = tid; x < 1024; x += 128) pl[x] = plan[(long)b * 1024 + x];
  for (int c = 0; c < 32; ++c)
    cn[c * D_ + tid] = (c < NPG_) ? h[((long)(2 * b + 1) * NPG_ + c) * D_ + tid] : 0.f;
  __syncthreads();
  float lsum = 0.f;
  for (int q = 0; q < 32; ++q) {
    float a = 0.f;
#pragma unroll 8
    for (int c = 0; c < 32; ++c) a += pl[q * 32 + c] * cn[c * D_ + tid];
    const float qv = (q < NPG_) ? h[((long)(2 * b) * NPG_ + q) * D_ + tid] : 0.f;
    lsum += fmaxf(qv - a, 0.f);
  }
  const float tot = block_reduce_sum(lsum, red);
  if (tid == 0) out[b] = -tot;
}

// edge_scores[b,i,j] = T[qf_i,cf_j]*T[qt_i,ct_j] + T[qf_i,ct_j]*T[qt_i,cf_j]
__global__ __launch_bounds__(256) void kron_k(
    const float* __restrict__ plan, const int* __restrict__ from,
    const int* __restrict__ to, float* __restrict__ se) {
  const int b = blockIdx.x, tid = threadIdx.x;
  __shared__ float pl[1024];
  __shared__ int qf[64], qt[64], cf[64], ct[64];
  for (int x = tid; x < 1024; x += 256) pl[x] = plan[(long)b * 1024 + x];
  if (tid < 64) {
    qf[tid] = from[(long)(2 * b) * EPG_ + tid] % NPG_;
    qt[tid] = to  [(long)(2 * b) * EPG_ + tid] % NPG_;
    cf[tid] = from[(long)(2 * b + 1) * EPG_ + tid] % NPG_;
    ct[tid] = to  [(long)(2 * b + 1) * EPG_ + tid] % NPG_;
  }
  __syncthreads();
  for (int x = tid; x < 4096; x += 256) {
    int i = x >> 6, j = x & 63;
    float s = pl[qf[i] * 32 + cf[j]] * pl[qt[i] * 32 + ct[j]] +
              pl[qf[i] * 32 + ct[j]] * pl[qt[i] * 32 + cf[j]];
    se[(long)b * 4096 + x] = s;
  }
}

// Sinkhorn on 64x64 (padded LDS rows) + edge alignment; out[b] -= CW*sum
__global__ __launch_bounds__(128) void edge_plan_align_k(
    const float* __restrict__ se, const float* __restrict__ em,
    float* __restrict__ out) {
  const int b = blockIdx.x, tid = threadIdx.x;
  __shared__ float la[64 * 65];
  __shared__ float ce[64 * D_];
  __shared__ float red[4];
  for (int x = tid; x < 4096; x += 128) {
    int i = x >> 6, j = x & 63;
    la[i * 65 + j] = se[(long)b * 4096 + x] * (1.f / TEMP_);
  }
  for (int j = 0; j < 64; ++j)
    ce[j * D_ + tid] = em[((long)(2 * b + 1) * EPG_ + j) * D_ + tid];
  __syncthreads();
  for (int it = 0; it < 20; ++it) {
    if (tid < 64) {                               // rows (axis=2)
      float m = -1e30f;
      for (int j = 0; j < 64; ++j) m = fmaxf(m, la[tid * 65 + j]);
      float s = 0.f;
      for (int j = 0; j < 64; ++j) s += expf(la[tid * 65 + j] - m);
      const float l = m + logf(s);
      for (int j = 0; j < 64; ++j) la[tid * 65 + j] -= l;
    }
    __syncthreads();
    if (tid < 64) {                               // cols (axis=1)
      float m = -1e30f;
      for (int i = 0; i < 64; ++i) m = fmaxf(m, la[i * 65 + tid]);
      float s = 0.f;
      for (int i = 0; i < 64; ++i) s += expf(la[i * 65 + tid] - m);
      const float l = m + logf(s);
      for (int i = 0; i < 64; ++i) la[i * 65 + tid] -= l;
    }
    __syncthreads();
  }
  for (int x = tid; x < 4096; x += 128) {
    int i = x >> 6, j = x & 63;
    la[i * 65 + j] = expf(la[i * 65 + j]);
  }
  __syncthreads();
  float lsum = 0.f;
  for (int i = 0; i < 64; ++i) {
    float a = 0.f;
#pragma unroll 8
    for (int j = 0; j < 64; ++j) a += la[i * 65 + j] * ce[j * D_ + tid];
    const float qv = em[((long)(2 * b) * EPG_ + i) * D_ + tid];
    lsum += fmaxf(qv - a, 0.f);
  }
  const float tot = block_reduce_sum(lsum, red);
  if (tid == 0) out[b] -= CW_ * tot;
}

// ---------------------------------------------------------------------------
// Host launcher
// ---------------------------------------------------------------------------
extern "C" void kernel_launch(void* const* d_in, const int* in_sizes, int n_in,
                              void* d_out, int out_size, void* d_ws, size_t ws_size,
                              hipStream_t stream) {
  const float* nf      = (const float*)d_in[0];
  const float* ef      = (const float*)d_in[1];
  const float* enc_nw  = (const float*)d_in[2];
  const float* enc_nb  = (const float*)d_in[3];
  const float* enc_ew  = (const float*)d_in[4];
  const float* enc_eb  = (const float*)d_in[5];
  const float* msg_w1  = (const float*)d_in[6];
  const float* msg_b1  = (const float*)d_in[7];
  const float* msg_w2  = (const float*)d_in[8];
  const float* msg_b2  = (const float*)d_in[9];
  const float* rmsg_w1 = (const float*)d_in[10];
  const float* rmsg_b1 = (const float*)d_in[11];
  const float* rmsg_w2 = (const float*)d_in[12];
  const float* rmsg_b2 = (const float*)d_in[13];
  const float* upd_w1  = (const float*)d_in[14];
  const float* upd_b1  = (const float*)d_in[15];
  const float* upd_w2  = (const float*)d_in[16];
  const float* upd_b2  = (const float*)d_in[17];
  const float* sk_w1   = (const float*)d_in[18];
  const float* sk_b1   = (const float*)d_in[19];
  const float* sk_w2   = (const float*)d_in[20];
  const float* sk_b2   = (const float*)d_in[21];
  const int*   from    = (const int*)d_in[22];
  const int*   to      = (const int*)d_in[23];
  const int*   qs      = (const int*)d_in[24];
  const int*   cs      = (const int*)d_in[25];
  float* out = (float*)d_out;

  // ---- workspace carve-out ----
  char* ws = (char*)d_ws;
  size_t off = 0;
  auto alloc = [&](size_t bytes) -> void* {
    void* p = ws + off;
    off = (off + bytes + 255) & ~(size_t)255;
    return p;
  };
  bf16_t* wb_enc_n = (bf16_t*)alloc(32 * 128 * 2);
  bf16_t* wb_enc_e = (bf16_t*)alloc(32 * 128 * 2);
  bf16_t* wb_msg1  = (bf16_t*)alloc(384 * 256 * 2);
  bf16_t* wb_msg2  = (bf16_t*)alloc(256 * 128 * 2);
  bf16_t* wb_rmsg1 = (bf16_t*)alloc(384 * 256 * 2);
  bf16_t* wb_rmsg2 = (bf16_t*)alloc(256 * 128 * 2);
  bf16_t* wb_upd1  = (bf16_t*)alloc(256 * 256 * 2);
  bf16_t* wb_upd2  = (bf16_t*)alloc(256 * 128 * 2);
  bf16_t* wb_sk1   = (bf16_t*)alloc(128 * 64 * 2);
  bf16_t* wb_sk2   = (bf16_t*)alloc(64 * 64 * 2);
  float*  h      = (float*) alloc(NN_ * D_ * 4);
  bf16_t* h_bf   = (bf16_t*)alloc(NN_ * D_ * 2);
  bf16_t* e_bf   = (bf16_t*)alloc(EE_ * D_ * 2);
  bf16_t* hid_bf = (bf16_t*)alloc(EE_ * 256 * 2);
  float*  fbuf   = (float*) alloc(EE_ * D_ * 4);
  char*   rb     = (char*)  alloc(EE_ * D_ * 4);   // rbuf; reused post-steps
  bf16_t* aggb   = (bf16_t*)alloc(NN_ * D_ * 2);
  float*  rbuf   = (float*)rb;
  // reuse rbuf region after the 3 propagation steps:
  bf16_t* hs_bf    = (bf16_t*)(rb);
  float*  tbuf     = (float*) (rb + 4194304);
  float*  scores_n = (float*) (rb + 8388608);
  float*  plan_n   = (float*) (rb + 9437184);
  float*  scores_e = (float*) (rb + 10485760);

  // ---- weight prep ----
  auto prep = [&](const float* s, bf16_t* d, int K, int Nc) {
    int total = K * Nc;
    prep_w<<<(total + 255) / 256, 256, 0, stream>>>(s, d, K, Nc);
  };
  prep(enc_nw, wb_enc_n, 32, 128);
  prep(enc_ew, wb_enc_e, 32, 128);
  prep(msg_w1, wb_msg1, 384, 256);
  prep(msg_w2, wb_msg2, 256, 128);
  prep(rmsg_w1, wb_rmsg1, 384, 256);
  prep(rmsg_w2, wb_rmsg2, 256, 128);
  prep(upd_w1, wb_upd1, 256, 256);
  prep(upd_w2, wb_upd2, 256, 128);
  prep(sk_w1, wb_sk1, 128, 64);
  prep(sk_w2, wb_sk2, 64, 64);

  const int GN = (int)(NN_ / 64);   // 224 blocks (4 waves x 16 rows)
  const int GE = (int)(EE_ / 64);   // 512
  const int GS = 16384 / 64;        // 256

  // ---- encoders ----
  gemm_k<0, 4, 0, 1><<<GN, 128, 0, stream>>>(nf, nullptr, nullptr, nullptr, nullptr,
      wb_enc_n, enc_nb, h, h_bf, 128);
  gemm_k<0, 1, 0, 1><<<GE, 128, 0, stream>>>(ef, nullptr, nullptr, nullptr, nullptr,
      wb_enc_e, enc_eb, nullptr, e_bf, 128);

  auto msg_pass = [&](const bf16_t* w1, const float* b1, const bf16_t* w2,
                      const float* b2, const int* ia, const int* ib,
                      float* dst, bool accum) {
    gemm_k<2, 1, 1, 12><<<GE, 128, 0, stream>>>(nullptr, h_bf, e_bf, ia, ib,
        w1, b1, nullptr, hid_bf, 256);
    if (accum)
      gemm_k<1, 2, 0, 8><<<GE, 128, 0, stream>>>(nullptr, hid_bf, nullptr, nullptr,
          nullptr, w2, b2, dst, nullptr, 128);
    else
      gemm_k<1, 0, 0, 8><<<GE, 128, 0, stream>>>(nullptr, hid_bf, nullptr, nullptr,
          nullptr, w2, b2, dst, nullptr, 128);
  };

  // ---- 3 propagation steps ----
  for (int s = 0; s < 3; ++s) {
    msg_pass(wb_msg1, msg_b1, wb_msg2, msg_b2, from, to, fbuf, false);
    msg_pass(wb_rmsg1, rmsg_b1, wb_rmsg2, rmsg_b2, to, from, rbuf, false);
    aggregate_k<<<(int)G_, 128, 0, stream>>>(fbuf, rbuf, from, to, aggb);
    gemm_k<3, 1, 1, 8><<<GN, 128, 0, stream>>>(nullptr, aggb, h_bf, nullptr, nullptr,
        wb_upd1, upd_b1, nullptr, hid_bf, 256);
    gemm_k<1, 3, 0, 8><<<GN, 128, 0, stream>>>(nullptr, hid_bf, nullptr, nullptr,
        nullptr, wb_upd2, upd_b2, h, h_bf, 128);
  }

  // ---- final messages: em = f + r (accumulated into fbuf) ----
  msg_pass(wb_msg1, msg_b1, wb_msg2, msg_b2, from, to, fbuf, false);
  msg_pass(wb_rmsg1, rmsg_b1, wb_rmsg2, rmsg_b2, to, from, fbuf, true);

  // ---- sinkhorn MLP on padded nodes ----
  pad_k<<<16384, 128, 0, stream>>>(h_bf, hs_bf);
  gemm_k<1, 1, 1, 4><<<GS, 128, 0, stream>>>(nullptr, hs_bf, nullptr, nullptr,
      nullptr, wb_sk1, sk_b1, nullptr, hid_bf, 64);
  gemm_k<1, 0, 0, 2><<<GS, 128, 0, stream>>>(nullptr, hid_bf, nullptr, nullptr,
      nullptr, wb_sk2, sk_b2, tbuf, nullptr, 64);

  // ---- plans + losses ----
  scores_node_k<<<B_, 256, 0, stream>>>(tbuf, qs, cs, scores_n);
  sinkhorn_node_k<<<B_, 32, 0, stream>>>(scores_n, plan_n);
  node_align_k<<<B_, 128, 0, stream>>>(plan_n, h, out);
  kron_k<<<B_, 256, 0, stream>>>(plan_n, from, to, scores_e);
  edge_plan_align_k<<<B_, 128, 0, stream>>>(scores_e, fbuf /* em */, out);

  (void)in_sizes; (void)n_in; (void)out_size; (void)ws_size;
}
